// MultiplexImageDecoder_14027363188963
// MI455X (gfx1250) — compile-verified, matrix operand order checked by reading, not run on hardware
//
#include <hip/hip_runtime.h>
#include <hip/hip_bf16.h>
#include <math.h>

// ---------------------------------------------------------------------------
// Types for gfx1250 WMMA (wave32)
// ---------------------------------------------------------------------------
typedef __bf16 bf16_t;
typedef __attribute__((ext_vector_type(16))) __bf16 v16bf;
typedef __attribute__((ext_vector_type(8)))  __bf16 v8bf;
typedef __attribute__((ext_vector_type(8)))  float  v8f;

// Problem constants (from reference)
#define NBATCH 64      // B*C
#define EDIM   256     // E == I
#define HIDDIM 1024    // 4*E
#define NPIX   1024    // H*W = 32*32

// GEMM tiling
#define BM 128
#define BN 128
#define BK 32
#define LDT 40         // padded LDS row stride (bf16 elems); 80B, 16B-aligned runs

// ---------------------------------------------------------------------------
// Fragment loader: per-lane 16 bf16 values for 16x32 (A) / 32x16 (B) operand.
// LDS layout: [row][k], stride LDT.  Lane l: row = l&15, koff = (l<16)?0:8;
// elems 0..7 -> k=koff..koff+7 ; elems 8..15 -> k=koff+16..koff+23.
// ---------------------------------------------------------------------------
__device__ __forceinline__ v16bf load_frag(const bf16_t* rowbase, int lane) {
    const int koff = (lane < 16) ? 0 : 8;
    v8bf lo = *(const v8bf*)(rowbase + koff);
    v8bf hi = *(const v8bf*)(rowbase + koff + 16);
    v16bf r;
#pragma unroll
    for (int i = 0; i < 8; ++i) { r[i] = lo[i]; r[i + 8] = hi[i]; }
    return r;
}

// ---------------------------------------------------------------------------
// Generic batched GEMM:  out[n][m][p] = act( A[m][k] * B[n][k][p] + bias[m] )
//   A: bf16 weights, row-major [M][K] (pre-converted / pre-transposed)
//   B: f32 or bf16 activations, [batch/bDiv][K][NPIX]
//   EPI: 0 = bias -> f32 ; 1 = bias+GELU -> bf16 ; 2 = bias+residual -> f32
//   gidx: optional per-batch gather (hyperkernel embedding lookup)
// grid = (NPIX/BN, ceil(M/BM), NBATCH), block = 256 (8 waves)
// ---------------------------------------------------------------------------
template <typename TB, int EPI>
__global__ __launch_bounds__(256) void gemm_bf16_wmma(
    const bf16_t* __restrict__ A, const TB* __restrict__ Bb,
    const float* __restrict__ bias, const float* __restrict__ res,
    float* __restrict__ outF, bf16_t* __restrict__ outBf,
    const int* __restrict__ gidx, long aGather, long biasGather,
    int M, int K, int bDiv)
{
    __shared__ __align__(16) bf16_t As[BM * LDT];   // [m][k]
    __shared__ __align__(16) bf16_t Bs[BN * LDT];   // [n][k]

    const int t     = threadIdx.x;
    const int lane  = t & 31;
    const int warp  = t >> 5;
    const int ntile = blockIdx.x * BN;
    const int mtile = blockIdx.y * BM;
    const int batch = blockIdx.z;

    const bf16_t* Ab = A + (gidx ? (long)gidx[batch] * aGather : 0);
    const TB*     Bp = Bb + (long)(batch / bDiv) * (long)K * NPIX;
    const float*  bp = bias + (gidx ? (long)gidx[batch] * biasGather : 0);

    const int m_warp = (warp >> 1) * 32;   // 4 warps along M
    const int n_warp = (warp & 1) * 64;    // 2 warps along N

    // staging assignment: each thread owns 16 k-consecutive elems of one row
    const int row   = t >> 1;              // 0..127 (m for A, n for B)
    const int kbase = (t & 1) * 16;        // 0 or 16

    v8f zero = {};
    v8f acc[2][4];
#pragma unroll
    for (int i = 0; i < 2; ++i)
#pragma unroll
        for (int j = 0; j < 4; ++j) acc[i][j] = zero;

    for (int k0 = 0; k0 < K; k0 += BK) {
        // ---- stage A tile: two global b128 loads -> two ds_store_b128 ----
        {
            const int gm = mtile + row;
            v8bf a0 = {}, a1 = {};
            if (gm < M) {                         // single hoisted guard
                const bf16_t* as = Ab + (long)gm * K + k0 + kbase;
                a0 = *(const v8bf*)as;
                a1 = *(const v8bf*)(as + 8);
            }
            *(v8bf*)(&As[row * LDT + kbase])     = a0;
            *(v8bf*)(&As[row * LDT + kbase + 8]) = a1;
        }
        // ---- stage B tile (transpose [k][n] -> [n][k]) ----
        {
            const TB* bs = Bp + (long)(k0 + kbase) * NPIX + ntile + row;
            v8bf b0, b1;
            if constexpr (sizeof(TB) == 4) {
#pragma unroll
                for (int i = 0; i < 8; ++i) {
                    b0[i] = (bf16_t)bs[(long)i * NPIX];
                    b1[i] = (bf16_t)bs[(long)(i + 8) * NPIX];
                }
            } else {
#pragma unroll
                for (int i = 0; i < 8; ++i) {
                    b0[i] = bs[(long)i * NPIX];
                    b1[i] = bs[(long)(i + 8) * NPIX];
                }
            }
            *(v8bf*)(&Bs[row * LDT + kbase])     = b0;
            *(v8bf*)(&Bs[row * LDT + kbase + 8]) = b1;
        }
        // prefetch next K-tile while this one is consumed
        if (k0 + BK < K) {
            __builtin_prefetch(Ab + (long)(mtile + row) * K + k0 + BK + kbase, 0, 3);
            __builtin_prefetch(Bp + (long)(k0 + BK + kbase) * NPIX + ntile + row, 0, 3);
        }
        __syncthreads();

        // ---- WMMA: 2x4 tiles of 16x16, K-step 32 ----
        const int lr = lane & 15;
        v16bf af[2], bfrag[4];
#pragma unroll
        for (int i = 0; i < 2; ++i)
            af[i] = load_frag(&As[(m_warp + 16 * i + lr) * LDT], lane);
#pragma unroll
        for (int j = 0; j < 4; ++j)
            bfrag[j] = load_frag(&Bs[(n_warp + 16 * j + lr) * LDT], lane);
#pragma unroll
        for (int i = 0; i < 2; ++i)
#pragma unroll
            for (int j = 0; j < 4; ++j)
                acc[i][j] = __builtin_amdgcn_wmma_f32_16x16x32_bf16(
                    false, af[i], false, bfrag[j], (short)0, acc[i][j],
                    false, false);
        __syncthreads();
    }

    // ---- epilogue (C/D layout: n = lane&15, m = r + 8*(lane>=16)) ----
    const int lr   = lane & 15;
    const int rowb = mtile + m_warp + ((lane >> 4) << 3);
    const int colb = ntile + n_warp + lr;
    const long ob  = (long)batch * M * NPIX;
#pragma unroll
    for (int i = 0; i < 2; ++i) {
#pragma unroll
        for (int j = 0; j < 4; ++j) {
#pragma unroll
            for (int r = 0; r < 8; ++r) {
                int rrow = rowb + 16 * i + r;
                int col  = colb + 16 * j;
                if (rrow < M) {
                    float v = acc[i][j][r] + bp[rrow];
                    long oi = ob + (long)rrow * NPIX + col;
                    if constexpr (EPI == 1) {
                        float gl = 0.5f * v * (1.0f + erff(v * 0.70710678118f));
                        outBf[oi] = (bf16_t)gl;
                    } else if constexpr (EPI == 2) {
                        outF[oi] = v + res[oi];
                    } else {
                        outF[oi] = v;
                    }
                }
            }
        }
    }
}

// ---------------------------------------------------------------------------
// Prep: elementwise f32 -> bf16 (8 elems / thread, vectorized)
// ---------------------------------------------------------------------------
__global__ __launch_bounds__(256) void cvt_f32_bf16(
    const float* __restrict__ in, bf16_t* __restrict__ out, long n8)
{
    const long i = (long)blockIdx.x * 256 + threadIdx.x;
    if (i >= n8) return;
    const float4* s = (const float4*)(in + i * 8);
    float4 v0 = s[0], v1 = s[1];
    v8bf o;
    o[0] = (bf16_t)v0.x; o[1] = (bf16_t)v0.y; o[2] = (bf16_t)v0.z; o[3] = (bf16_t)v0.w;
    o[4] = (bf16_t)v1.x; o[5] = (bf16_t)v1.y; o[6] = (bf16_t)v1.z; o[7] = (bf16_t)v1.w;
    *(v8bf*)(out + i * 8) = o;
}

// ---------------------------------------------------------------------------
// Prep: hk_w[ch][i][e] -> hk_wt[ch][e][i]  (256x256 per channel, bf16 out)
// grid = (8, 8, 64); block = 256 (32x8)
// ---------------------------------------------------------------------------
__global__ __launch_bounds__(256) void transpose_cvt_hk(
    const float* __restrict__ in, bf16_t* __restrict__ out)
{
    __shared__ float tile[32][33];
    const int ch = blockIdx.z;
    const int i0 = blockIdx.x * 32, e0 = blockIdx.y * 32;
    const int tx = threadIdx.x & 31, ty = threadIdx.x >> 5;
    const float* src = in + ((long)ch * EDIM + i0) * EDIM + e0;
#pragma unroll
    for (int r = ty; r < 32; r += 8) tile[r][tx] = src[(long)r * EDIM + tx];
    __syncthreads();
    bf16_t* dst = out + ((long)ch * EDIM + e0) * EDIM + i0;
#pragma unroll
    for (int r = ty; r < 32; r += 8) dst[(long)r * EDIM + tx] = (bf16_t)tile[tx][r];
}

// ---------------------------------------------------------------------------
// Depthwise 7x7 conv, pad 3.  One block per (n, e); 38x38 zero-haloed tile.
// ---------------------------------------------------------------------------
__global__ __launch_bounds__(256) void dwconv7(
    const float* __restrict__ in, float* __restrict__ out,
    const float* __restrict__ w, const float* __restrict__ b)
{
    const int n = blockIdx.x >> 8;
    const int e = blockIdx.x & 255;
    __shared__ float tile[38 * 38];
    __shared__ float wt[49];
    const int t = threadIdx.x;
    if (t < 49) wt[t] = w[e * 49 + t];
    const long base = ((long)n * EDIM + e) * NPIX;
    for (int i = t; i < 38 * 38; i += 256) {
        int y = i / 38 - 3, x = i % 38 - 3;
        float v = 0.f;
        if ((unsigned)y < 32u && (unsigned)x < 32u) v = in[base + y * 32 + x];
        tile[i] = v;
    }
    __syncthreads();
    const float bb = b[e];
    for (int p = t; p < NPIX; p += 256) {
        int y = p >> 5, x = p & 31;
        float s = bb;
#pragma unroll
        for (int dy = 0; dy < 7; ++dy)
#pragma unroll
            for (int dx = 0; dx < 7; ++dx)
                s += tile[(y + dy) * 38 + x + dx] * wt[dy * 7 + dx];
        out[base + p] = s;
    }
}

// ---------------------------------------------------------------------------
// LayerNorm over channels per pixel; writes bf16 for fc1's B operand.
// ---------------------------------------------------------------------------
__global__ __launch_bounds__(256) void layernorm_k(
    const float* __restrict__ in, bf16_t* __restrict__ out,
    const float* __restrict__ g, const float* __restrict__ b)
{
    const int tid = blockIdx.x * 256 + threadIdx.x;       // 65536 pixels
    const long base = (long)(tid >> 10) * EDIM * NPIX + (tid & 1023);
    float s = 0.f, ss = 0.f;
    for (int e = 0; e < EDIM; ++e) {
        float v = in[base + (long)e * NPIX];
        s += v; ss += v * v;
    }
    const float mu = s * (1.f / EDIM);
    const float var = ss * (1.f / EDIM) - mu * mu;
    const float rstd = rsqrtf(var + 1e-5f);
    for (int e = 0; e < EDIM; ++e) {
        float v = in[base + (long)e * NPIX];
        out[base + (long)e * NPIX] = (bf16_t)((v - mu) * rstd * g[e] + b[e]);
    }
}

// ---------------------------------------------------------------------------
// GRN: gx[n][d] = sqrt(sum_p x^2)  (one wave32 per (n,d) row; v8bf loads)
// ---------------------------------------------------------------------------
__global__ __launch_bounds__(256) void grn_stats(
    const bf16_t* __restrict__ g, float* __restrict__ gx)
{
    const int lane = threadIdx.x & 31;
    const int row  = blockIdx.x * 8 + (threadIdx.x >> 5);  // 65536 rows
    const bf16_t* p = g + (long)row * NPIX + lane * 32;    // 32 contiguous/lane
    float s = 0.f;
#pragma unroll
    for (int q = 0; q < 4; ++q) {
        v8bf v = *(const v8bf*)(p + q * 8);
#pragma unroll
        for (int j = 0; j < 8; ++j) { float f = (float)v[j]; s += f * f; }
    }
#pragma unroll
    for (int off = 16; off; off >>= 1) s += __shfl_xor(s, off, 32);
    if (lane == 0) gx[row] = sqrtf(s);
}

// nx[n][d] = gx[n][d] / (mean_d gx[n][:] + eps) ; one block per image n
__global__ __launch_bounds__(256) void grn_mean(
    const float* __restrict__ gx, float* __restrict__ nx)
{
    const int n = blockIdx.x;
    __shared__ float red[256];
    float s = 0.f;
    for (int d = threadIdx.x; d < HIDDIM; d += 256) s += gx[n * HIDDIM + d];
    red[threadIdx.x] = s;
    __syncthreads();
    for (int o = 128; o; o >>= 1) {
        if (threadIdx.x < o) red[threadIdx.x] += red[threadIdx.x + o];
        __syncthreads();
    }
    const float inv = 1.f / (red[0] * (1.f / HIDDIM) + 1e-6f);
    for (int d = threadIdx.x; d < HIDDIM; d += 256)
        nx[n * HIDDIM + d] = gx[n * HIDDIM + d] * inv;
}

// out = gamma[d]*(x*nx[n,d]) + beta[d] + x   (in-place, 8 bf16 per thread)
__global__ __launch_bounds__(256) void grn_apply(
    bf16_t* __restrict__ g, const float* __restrict__ nx,
    const float* __restrict__ gamma, const float* __restrict__ beta)
{
    const long base = ((long)blockIdx.x * 256 + threadIdx.x) * 8;  // 8 | 1024
    const int nd = (int)(base >> 10);
    const int d  = nd & 1023;
    const float nxv = nx[nd], ga = gamma[d], be = beta[d];
    v8bf v = *(v8bf*)(g + base);
#pragma unroll
    for (int j = 0; j < 8; ++j) {
        float x = (float)v[j];
        v[j] = (bf16_t)(ga * (x * nxv) + be + x);
    }
    *(v8bf*)(g + base) = v;
}

// ---------------------------------------------------------------------------
// Pixel-shuffle: each thread emits the 8 contiguous (s2,o) outputs for one
// (n, h, s1, w):  out[n][h*4+s1][w*4+s2][o] = pred[n][s1*8+s2*2+o][h][w]
// ---------------------------------------------------------------------------
__global__ __launch_bounds__(256) void pixshuf(
    const float* __restrict__ p, float* __restrict__ out)
{
    const int i = blockIdx.x * 256 + threadIdx.x;   // 64*32*4*32 = 262144
    const int n  = i >> 12;
    const int r  = i & 4095;
    const int h  = r >> 7;
    const int s1 = (r >> 5) & 3;
    const int w  = r & 31;
    const int pp = h * 32 + w;
    const float* pb = p + ((long)n * 32 + s1 * 8) * NPIX + pp;
    float vals[8];
#pragma unroll
    for (int q = 0; q < 8; ++q) vals[q] = pb[(long)q * NPIX];
    const long ob = ((long)(n * 128 + h * 4 + s1) * 128 + w * 4) * 2;
    *(float4*)(out + ob)     = make_float4(vals[0], vals[1], vals[2], vals[3]);
    *(float4*)(out + ob + 4) = make_float4(vals[4], vals[5], vals[6], vals[7]);
}

// ---------------------------------------------------------------------------
// Host launcher
// ---------------------------------------------------------------------------
extern "C" void kernel_launch(void* const* d_in, const int* in_sizes, int n_in,
                              void* d_out, int out_size, void* d_ws, size_t ws_size,
                              hipStream_t stream)
{
    const float* x       = (const float*)d_in[0];   // [4,256,32,32]
    const int*   indices = (const int*)  d_in[1];   // [64]
    const float* hk_w    = (const float*)d_in[2];   // [64, 256*256]
    const float* hk_b    = (const float*)d_in[3];   // [64, 256]
    const float* dw_w    = (const float*)d_in[4];   // [4,256,1,7,7]
    const float* dw_b    = (const float*)d_in[5];   // [4,256]
    const float* ln_g    = (const float*)d_in[6];   // [4,256]
    const float* ln_b    = (const float*)d_in[7];   // [4,256]
    const float* fc1_w   = (const float*)d_in[8];   // [4,1024,256]
    const float* fc1_b   = (const float*)d_in[9];   // [4,1024]
    const float* grn_g   = (const float*)d_in[10];  // [4,1024]
    const float* grn_b   = (const float*)d_in[11];  // [4,1024]
    const float* fc2_w   = (const float*)d_in[12];  // [4,256,1024]
    const float* fc2_b   = (const float*)d_in[13];  // [4,256]
    const float* pred_w  = (const float*)d_in[14];  // [32,256]
    const float* pred_b  = (const float*)d_in[15];  // [32]
    float* out = (float*)d_out;

    // workspace layout (bytes)
    char* ws = (char*)d_ws;
    float*  h      = (float*)(ws + 0);            //  64 MB: activations [64][256][1024]
    float*  tb     = (float*)(ws + 67108864);     //  64 MB: dwconv out
    bf16_t* lnb    = (bf16_t*)(ws + 134217728);   //  32 MB: LN out (bf16)
    bf16_t* g      = (bf16_t*)(ws + 167772160);   // 128 MB: fc1/GRN act (bf16)
    float*  gx     = (float*)(ws + 301989888);    // 256 KB
    float*  nx     = (float*)(ws + 302252032);    // 256 KB
    float*  pr     = (float*)(ws + 302514176);    //   8 MB: pred out [64][32][1024]
    bf16_t* hk_wt  = (bf16_t*)(ws + 310902784);   //   8 MB: hk_w^T bf16 [64][E][I]
    bf16_t* fc1_wb = (bf16_t*)(ws + 319291392);   //   2 MB
    bf16_t* fc2_wb = (bf16_t*)(ws + 321388544);   //   2 MB
    bf16_t* pred_wb= (bf16_t*)(ws + 323485696);   //  16 KB

    // 0) weight prep: bf16 conversion (+ transpose for hyperkernel table)
    transpose_cvt_hk<<<dim3(8, 8, 64), 256, 0, stream>>>(hk_w, hk_wt);
    cvt_f32_bf16<<<512, 256, 0, stream>>>(fc1_w, fc1_wb, 131072);   // 4*1024*256/8
    cvt_f32_bf16<<<512, 256, 0, stream>>>(fc2_w, fc2_wb, 131072);   // 4*256*1024/8
    cvt_f32_bf16<<<4,   256, 0, stream>>>(pred_w, pred_wb, 1024);   // 32*256/8

    // 1) Hyperkernel GEMM: h[n][e][p] = sum_i hk_wt[idx[n]][e][i]*x[b][i][p] + hk_b[idx[n]][e]
    gemm_bf16_wmma<float, 0><<<dim3(NPIX / BN, EDIM / BM, NBATCH), 256, 0, stream>>>(
        hk_wt, x, hk_b, nullptr, h, nullptr, indices,
        (long)EDIM * EDIM, (long)EDIM, EDIM, EDIM, /*bDiv=*/16);

    // 2) Four ConvNeXt blocks
    for (int l = 0; l < 4; ++l) {
        dwconv7<<<NBATCH * EDIM, 256, 0, stream>>>(h, tb, dw_w + l * EDIM * 49, dw_b + l * EDIM);
        layernorm_k<<<(NBATCH * NPIX) / 256, 256, 0, stream>>>(tb, lnb, ln_g + l * EDIM, ln_b + l * EDIM);
        // fc1 + GELU -> bf16
        gemm_bf16_wmma<bf16_t, 1><<<dim3(NPIX / BN, HIDDIM / BM, NBATCH), 256, 0, stream>>>(
            fc1_wb + (long)l * HIDDIM * EDIM, lnb, fc1_b + l * HIDDIM, nullptr,
            nullptr, g, nullptr, 0, 0, HIDDIM, EDIM, 1);
        // GRN
        grn_stats<<<(NBATCH * HIDDIM) / 8, 256, 0, stream>>>(g, gx);
        grn_mean<<<NBATCH, 256, 0, stream>>>(gx, nx);
        grn_apply<<<(unsigned)(((long)NBATCH * HIDDIM * NPIX) / (256 * 8)), 256, 0, stream>>>(
            g, nx, grn_g + l * HIDDIM, grn_b + l * HIDDIM);
        // fc2 + residual -> f32 (in-place residual into h)
        gemm_bf16_wmma<bf16_t, 2><<<dim3(NPIX / BN, EDIM / BM, NBATCH), 256, 0, stream>>>(
            fc2_wb + (long)l * EDIM * HIDDIM, g, fc2_b + l * EDIM, h,
            h, nullptr, nullptr, 0, 0, EDIM, HIDDIM, 1);
    }

    // 3) pred 1x1 conv (M=32) + pixel shuffle
    gemm_bf16_wmma<float, 0><<<dim3(NPIX / BN, 1, NBATCH), 256, 0, stream>>>(
        pred_wb, h, pred_b, nullptr, pr, nullptr, nullptr, 0, 0, 32, EDIM, 1);
    pixshuf<<<(NBATCH * 32 * NPIX) / 256, 256, 0, stream>>>(pr, out);
}
